// CustomLinear_51110110822461
// MI455X (gfx1250) — matrix operand order repaired and checked by reading, not compiled, not run on hardware
//
#include <hip/hip_runtime.h>

// ---------------------------------------------------------------------------
// CustomLinear: out[m,n] = sum_k x[m,k] * w[n,k] + bias[n]
// M=8192, K(IN_F)=4096, N(OUT_F)=4096.
//
// MI455X plan:
//  - Compute-bound GEMM (AI ~860 FLOP/B) -> v_wmma_f32_16x16x32_bf16.
//  - One-shot f32->bf16 downconvert of x and w into d_ws (96 MB, L2-resident).
//  - 128x128 block tile, BK=32, 256 threads (8 wave32s, 2x4 wave grid),
//    each wave computes a 64x32 C tile = 8 WMMA accumulators.
//  - Tiles move global->LDS via the Tensor Data Mover: wave 0 issues the A
//    descriptor, wave 1 the B descriptor per K-step; TDM pad (16 B per 64 B
//    row) lands the tile at an 80 B conflict-free LDS stride. No staging
//    VGPRs -> no spills. Sync: s_wait_tensorcnt + 1 barrier per K-step.
//  - Fragment layouts follow CDNA5 ISA 7.12.2: A lane-half h reads
//    K = 8h..8h+7 and 16+8h..16+8h+7; B lane-half h reads the same K as A.
// ---------------------------------------------------------------------------

typedef __bf16 bf16_t;
typedef __attribute__((ext_vector_type(16))) __bf16  v16bf;
typedef __attribute__((ext_vector_type(8)))  __bf16  v8bf;
typedef __attribute__((ext_vector_type(8)))  float   v8f;
typedef __attribute__((ext_vector_type(4)))  float   v4f;
typedef unsigned int u32;
typedef __attribute__((ext_vector_type(4))) u32 u32x4;
typedef __attribute__((ext_vector_type(8))) u32 u32x8;

#define M_DIM 8192
#define K_DIM 4096
#define N_DIM 4096

#define BM 128
#define BN 128
#define BK 32
#define BKP 40   // LDS row stride in bf16: 64 B data + 16 B TDM pad = 80 B

// ------------------------------ f32 -> bf16 --------------------------------
__global__ __launch_bounds__(256) void cvt_f32_to_bf16_k(
    const float* __restrict__ src, bf16_t* __restrict__ dst, long n) {
  long i = ((long)blockIdx.x * blockDim.x + threadIdx.x) * 8;
  if (i + 8 <= n) {
    v4f a = *(const v4f*)(src + i);
    v4f b = *(const v4f*)(src + i + 4);
    v8bf o;
    o[0] = (__bf16)a[0]; o[1] = (__bf16)a[1];
    o[2] = (__bf16)a[2]; o[3] = (__bf16)a[3];
    o[4] = (__bf16)b[0]; o[5] = (__bf16)b[1];
    o[6] = (__bf16)b[2]; o[7] = (__bf16)b[3];
    *(v8bf*)(dst + i) = o;
  }
}

// Fragment load: two 16B-aligned ds_load_b128 from (possibly split) offsets.
__device__ __forceinline__ v16bf load_frag_2x16(const bf16_t* p_lo,
                                                const bf16_t* p_hi) {
  union { uint4 u[2]; v16bf v; } r;
  r.u[0] = *(const uint4*)(p_lo);
  r.u[1] = *(const uint4*)(p_hi);
  return r.v;
}

// --------------------- TDM: global -> LDS 2D tile load ---------------------
// Loads a 128-row x 32-bf16 tile (memory row stride = K_DIM elems) to LDS at
// lds_addr; pad_enable inserts 16 B after each 64 B row (pad_interval code 3
// = 16 DWORDs, pad_amount code 3 = 4 DWORDs) -> 80 B LDS row stride.
// D# packing per CDNA5 ISA. TENSORcnt.
__device__ __forceinline__ void tdm_load_tile(u32 lds_addr,
                                              const bf16_t* gptr,
                                              u32 tensor_rows) {
  const unsigned long long ga = (unsigned long long)(uintptr_t)gptr;
  u32x4 g0;
  g0.x = 1u;                                   // count=1, user descriptor
  g0.y = lds_addr;                             // LDS byte address
  g0.z = (u32)ga;                              // global_addr[31:0]
  g0.w = (u32)(ga >> 32) | (2u << 30);         // global_addr[56:32] | type=2

  u32x8 g1;
  g1[0] = (1u << 16)                           // data_size = 2 bytes
        | (1u << 20)                           // pad_enable
        | (3u << 22)                           // pad_interval = 16 DWORDs
        | (3u << 25);                          // pad_amount   = 4 DWORDs
  g1[1] = ((u32)K_DIM & 0xFFFFu) << 16;        // tensor_dim0[15:0] @ bit48
  g1[2] = ((u32)K_DIM >> 16)                   // tensor_dim0[31:16]
        | ((tensor_rows & 0xFFFFu) << 16);     // tensor_dim1[15:0]
  g1[3] = (tensor_rows >> 16)                  // tensor_dim1[31:16]
        | ((u32)BK << 16);                     // tile_dim0 = 32
  g1[4] = (u32)BM;                             // tile_dim1 = 128, tile_dim2=0
  g1[5] = (u32)K_DIM;                          // tensor_dim0_stride[31:0]
  g1[6] = 0u;                                  // stride[47:32] | dim1_stride
  g1[7] = 0u;

  asm volatile("tensor_load_to_lds %0, %1" :: "s"(g0), "s"(g1) : "memory");
}

// ------------------------------- WMMA GEMM ---------------------------------
__global__ __launch_bounds__(256) void gemm_bf16_wmma_k(
    const bf16_t* __restrict__ A,   // x_bf16,  M x K row-major
    const bf16_t* __restrict__ B,   // w_bf16,  N x K row-major
    const float*  __restrict__ bias,
    float*        __restrict__ C)   // out, M x N row-major
{
  __shared__ bf16_t As[2][BM][BKP];
  __shared__ bf16_t Bs[2][BN][BKP];

  const int t    = threadIdx.x;
  const int lane = t & 31;
  const int wave = t >> 5;      // 0..7
  const int wm   = wave >> 2;   // 0..1  (M direction)
  const int wn   = wave & 3;    // 0..3  (N direction)
  const int m0   = blockIdx.y * BM;
  const int n0   = blockIdx.x * BN;

  const int rl = lane & 15;           // row/col within 16x16 tile
  const int h8 = (lane >> 4) << 3;    // A K-base per lane half: 0 or 8
  const int kh = (lane >> 4) << 4;    // unused; 0 or 16

  const u32 ldsA0 = (u32)(uintptr_t)&As[0][0][0];
  const u32 ldsB0 = (u32)(uintptr_t)&Bs[0][0][0];
  const u32 bufBytes = (u32)(BM * BKP * sizeof(bf16_t));  // 10240 B

  v8f acc[4][2];
#pragma unroll
  for (int mt = 0; mt < 4; ++mt)
#pragma unroll
    for (int nt = 0; nt < 2; ++nt)
      acc[mt][nt] = (v8f){0.f, 0.f, 0.f, 0.f, 0.f, 0.f, 0.f, 0.f};

  const int KT = K_DIM / BK;  // 128 K-steps

  // ---- prologue: TDM-load K-step 0 into buffer 0 ----
  if (wave == 0)
    tdm_load_tile(ldsA0, A + (size_t)m0 * K_DIM, (u32)M_DIM);
  else if (wave == 1)
    tdm_load_tile(ldsB0, B + (size_t)n0 * K_DIM, (u32)N_DIM);
  __builtin_amdgcn_s_wait_tensorcnt(0);   // no-op for non-issuing waves
  __syncthreads();

  for (int kt = 0; kt < KT; ++kt) {
    const int cur = kt & 1;
    const int nxt = cur ^ 1;

    // kick off TDM for K-step kt+1 into the other buffer (waves 0 and 1)
    if (kt + 1 < KT) {
      const size_t koff = (size_t)(kt + 1) * BK;
      if (wave == 0)
        tdm_load_tile(ldsA0 + (u32)nxt * bufBytes,
                      A + (size_t)m0 * K_DIM + koff, (u32)M_DIM);
      else if (wave == 1)
        tdm_load_tile(ldsB0 + (u32)nxt * bufBytes,
                      B + (size_t)n0 * K_DIM + koff, (u32)N_DIM);
    }

    // ---- compute on buffer `cur`: fragment loads + 8 WMMAs ----
    // A fragment (16x32, CDNA5 ISA): lane half h holds K = 8h..8h+7 and
    // 16+8h..16+8h+7 -> two split 16 B loads.
    v16bf af[4], bfr[2];
#pragma unroll
    for (int mt = 0; mt < 4; ++mt) {
      const bf16_t* row = &As[cur][wm * 64 + mt * 16 + rl][0];
      af[mt] = load_frag_2x16(row + h8, row + 16 + h8);
    }
    // B fragment (32x16): lane half h holds K = 8h..8h+7 and 16+8h..16+8h+7.
#pragma unroll
    for (int nt = 0; nt < 2; ++nt) {
      const bf16_t* row = &Bs[cur][wn * 32 + nt * 16 + rl][0];
      bfr[nt] = load_frag_2x16(row + h8, row + 16 + h8);
    }

#pragma unroll
    for (int mt = 0; mt < 4; ++mt)
#pragma unroll
      for (int nt = 0; nt < 2; ++nt)
        acc[mt][nt] = __builtin_amdgcn_wmma_f32_16x16x32_bf16(
            /*neg_a=*/false, af[mt], /*neg_b=*/false, bfr[nt],
            /*c_mod=*/(short)0, acc[mt][nt],
            /*reuse_a=*/false, /*reuse_b=*/false);

    // drain TDM, then release everyone into the next K-step
    if (kt + 1 < KT) {
      __builtin_amdgcn_s_wait_tensorcnt(0);
      __syncthreads();
    }
  }

  // ---- epilogue: bias add + f32 store ----
  // C/D 16x16 f32 layout: VGPR r -> lanes 0-15: (M=r, N=lane),
  //                                  lanes 16-31: (M=r+8, N=lane-16)
#pragma unroll
  for (int nt = 0; nt < 2; ++nt) {
    const int n  = n0 + wn * 32 + nt * 16 + (lane & 15);
    const float bn = bias[n];
#pragma unroll
    for (int mt = 0; mt < 4; ++mt) {
      const int mbase = m0 + wm * 64 + mt * 16 + ((lane >> 4) << 3);
#pragma unroll
      for (int r = 0; r < 8; ++r)
        C[(size_t)(mbase + r) * N_DIM + n] = acc[mt][nt][r] + bn;
    }
  }
}

// ------------------------------- launcher ----------------------------------
extern "C" void kernel_launch(void* const* d_in, const int* in_sizes, int n_in,
                              void* d_out, int out_size, void* d_ws, size_t ws_size,
                              hipStream_t stream) {
  (void)in_sizes; (void)n_in; (void)out_size; (void)ws_size;

  const float* x    = (const float*)d_in[0];  // [M, K]
  const float* w    = (const float*)d_in[1];  // [N, K]
  const float* bias = (const float*)d_in[2];  // [N]
  float*       out  = (float*)d_out;          // [M, N]

  bf16_t* xb = (bf16_t*)d_ws;                                     // 64 MB
  bf16_t* wb = (bf16_t*)((char*)d_ws +
                         (size_t)M_DIM * K_DIM * sizeof(bf16_t)); // +32 MB

  {
    const long nx = (long)M_DIM * K_DIM;       // 33.5M elems, /8/256 exact
    cvt_f32_to_bf16_k<<<(unsigned)(nx / 8 / 256), 256, 0, stream>>>(x, xb, nx);
    const long nw = (long)N_DIM * K_DIM;       // 16.8M elems
    cvt_f32_to_bf16_k<<<(unsigned)(nw / 8 / 256), 256, 0, stream>>>(w, wb, nw);
  }

  dim3 grid(N_DIM / BN, M_DIM / BM);  // (32, 64)
  gemm_bf16_wmma_k<<<grid, 256, 0, stream>>>(xb, wb, bias, out);
}
